// SpikingNeuralNetwork_59648505807275
// MI455X (gfx1250) — matrix operand order repaired and checked by reading, not compiled, not run on hardware
//
#include <hip/hip_runtime.h>

// ---------------------------------------------------------------------------
// Fused spiking-network simulation for MI455X (gfx1250, wave32, WMMA).
//
// All matmul A-operands (spikes) are exactly {0,1} -> exact in f16.
// Weights are split W = hi + lo (both f16); A@hi + A@lo accumulated in the
// f32 WMMA accumulator reconstructs f32-level accuracy while running on
// v_wmma_f32_16x16x32_f16.
//
// One workgroup (8 wave32) owns 16 batch rows for all 100 time steps; state
// (s0,m0,s1,m1,rp) lives in VGPRs, spikes are exchanged through LDS.
// The next step's input tile is fetched with GLOBAL_LOAD_ASYNC_TO_LDS_B128
// (ASYNCcnt) into a double buffer while the current step's WMMAs execute.
// ---------------------------------------------------------------------------

typedef _Float16 h16v __attribute__((ext_vector_type(16)));
typedef float    f8v  __attribute__((ext_vector_type(8)));

#define B_TOT   512
#define T_STEPS 100
#define I_DIM   700
#define I_PAD   704
#define H_DIM   128
#define O_DIM   20
#define O_PAD   32
#define TILE_B  16

constexpr float ALPHA = 0.8187307530779818f;  // exp(-dt/tau_syn)
constexpr float BETA  = 0.9048374180359595f;  // exp(-dt/tau_mem)
constexpr float LAM   = 0.9512294245007140f;  // exp(-dt/tau_ro)

union Frag { h16v v; float4 f[2]; };

// A-matrix 16x32 f16 fragment (ISA 7.12.2): lane (l&15) = row M,
// lanes 0-15 hold K {0..7, 16..23}, lanes 16-31 hold K {8..15, 24..31}.
__device__ __forceinline__ h16v load_a_frag(const _Float16* __restrict__ base,
                                            int row, int stride, int kbase, int lhi) {
  const _Float16* p = base + row * stride + kbase + lhi * 8;
  Frag u;
  u.f[0] = *(const float4*)(p);        // K +0..7
  u.f[1] = *(const float4*)(p + 16);   // K +16..23
  return u.v;
}

// B-matrix 32x16 f16 fragment: lane (l&15) = column N; lanes 0-15 hold
// K {0..7, 16..23}, lanes 16-31 hold K {8..15, 24..31} (matches A). Since
// B = W^T and W is row-major (N x K), these are two 16-byte runs of row N of W.
__device__ __forceinline__ h16v load_b_frag(const _Float16* __restrict__ base,
                                            int row, int stride, int kbase, int lhi) {
  const _Float16* p = base + row * stride + kbase + lhi * 8;
  Frag u;
  u.f[0] = *(const float4*)(p);        // K +0..7
  u.f[1] = *(const float4*)(p + 16);   // K +16..23
  return u.v;
}

__device__ __forceinline__ f8v wmma_f16(h16v a, h16v b, f8v c) {
  return __builtin_amdgcn_wmma_f32_16x16x32_f16(
      /*neg_a=*/false, a, /*neg_b=*/false, b,
      /*c_mod=*/(short)0, c, /*reuse_a=*/false, /*reuse_b=*/false);
}

// Async global->LDS 16B copy (GV mode). VDST VGPR holds the wave-relative
// LDS byte offset; flat shared pointers carry that offset in their low 32
// bits (ISA 10.2). Tracked by ASYNCcnt.
__device__ __forceinline__ void async_copy_b128(unsigned ldsoff, const void* gptr) {
  asm volatile("global_load_async_to_lds_b128 %0, %1, off"
               :: "v"(ldsoff), "v"(gptr) : "memory");
}
__device__ __forceinline__ void wait_async0() {
  asm volatile("s_wait_asynccnt 0" ::: "memory");
}

// ---------------------------------------------------------------------------
// Prepass: split f32 weights into hi/lo f16 pair with zero padding.
// ---------------------------------------------------------------------------
__global__ void split_f16(const float* __restrict__ src,
                          _Float16* __restrict__ hi, _Float16* __restrict__ lo,
                          int rs, int cs, int rd, int cd) {
  int idx = blockIdx.x * 256 + threadIdx.x;
  if (idx >= rd * cd) return;
  int r = idx / cd, c = idx - r * cd;
  float w = (r < rs && c < cs) ? src[r * cs + c] : 0.0f;
  _Float16 h = (_Float16)w;
  hi[idx] = h;
  lo[idx] = (_Float16)(w - (float)h);
}

// ---------------------------------------------------------------------------
// Main fused kernel.
// ---------------------------------------------------------------------------
__global__ __launch_bounds__(256) void snn_fused(
    const float* __restrict__ x,                                  // (B,T,I) f32 spikes
    const _Float16* __restrict__ W0hi, const _Float16* __restrict__ W0lo,  // (H, I_PAD)
    const _Float16* __restrict__ W1hi, const _Float16* __restrict__ W1lo,  // (H, H)
    const _Float16* __restrict__ V1hi, const _Float16* __restrict__ V1lo,  // (H, H)
    const _Float16* __restrict__ Wrhi, const _Float16* __restrict__ Wrlo,  // (O_PAD, H)
    const float* __restrict__ br,
    float* __restrict__ out)                                      // (B,T,O) f32
{
  __shared__ float    xraw[2][TILE_B * I_PAD];    // 2x45 KB: async-staged raw x_t
  __shared__ _Float16 xbuf[TILE_B * I_PAD];       // 22.5 KB: x_t tile as f16
  __shared__ _Float16 k0buf[TILE_B * H_DIM];      //  4 KB: layer-0 spikes
  __shared__ _Float16 k1buf[2][TILE_B * H_DIM];   //  8 KB: layer-1 spikes (dbl buf)

  const int tid   = threadIdx.x;
  const int wid   = tid >> 5;        // wave id 0..7 (wave32)
  const int lane  = tid & 31;
  const int lrow  = lane & 15;
  const int lhi   = lane >> 4;
  const int nbase = wid * 16;        // this wave's H-column tile
  const int bbase = blockIdx.x * TILE_B;

  // Async staging: wave w owns batch rows 2w, 2w+1. Each row = 175 16-byte
  // chunks of f32; 350 chunks/wave -> 11 (partially masked) b128 issues.
  auto stage_async = [&](int t, int par) {
    const int rbase = wid * 2;
    float* dstbase = &xraw[par][0];
    #pragma unroll
    for (int inst = 0; inst < 11; ++inst) {
      int gi = inst * 32 + lane;                 // 0..351
      if (gi < 2 * (I_DIM / 4)) {                // 350 valid chunks
        int rsel = (gi >= (I_DIM / 4)) ? 1 : 0;
        int row  = rbase + rsel;
        int cc   = gi - rsel * (I_DIM / 4);      // 16B chunk within row
        const float* gp = x + ((size_t)(bbase + row) * T_STEPS + t) * I_DIM + cc * 4;
        unsigned loff = (unsigned)(uintptr_t)(dstbase + row * I_PAD + cc * 4);
        async_copy_b128(loff, gp);
      }
    }
  };

  // k1(t=-1) = 0
  for (int i = tid; i < TILE_B * H_DIM; i += 256) k1buf[0][i] = (_Float16)0.0f;

  // kick off x_0 staging
  stage_async(0, 0);

  f8v s0 = {}, m0 = {}, s1 = {}, m1 = {}, rp = {};
  float brv = 0.0f;
  if (wid < 2) {
    int o = nbase + lrow;
    brv = (o < O_DIM) ? br[o] : 0.0f;
  }

  const int xrow  = tid >> 4;        // 0..15: batch row this thread converts
  const int xcol0 = tid & 15;

  for (int t = 0; t < T_STEPS; ++t) {
    const int par = t & 1;

    // ---- wait for this step's async tile, then prefetch next step ----
    wait_async0();
    __syncthreads();
    if (t + 1 < T_STEPS) stage_async(t + 1, par ^ 1);   // hidden behind WMMAs

    // ---- convert x_t (f32 -> f16, zero-pad 700->704) ----
    {
      const float* srcp = &xraw[par][xrow * I_PAD];
      _Float16* dst = xbuf + xrow * I_PAD;
      for (int c = xcol0; c < I_PAD; c += 16) {
        float v = (c < I_DIM) ? srcp[c] : 0.0f;
        dst[c] = (_Float16)v;
      }
    }
    __syncthreads();

    // ---- layer 0: s0 = ALPHA*s0 + x_t @ W0^T (hi+lo split) ----
    {
      f8v acc = s0 * ALPHA;                     // leak folded into C operand
      #pragma unroll 2
      for (int kt = 0; kt < I_PAD / 32; ++kt) {
        int kb = kt * 32;
        h16v a  = load_a_frag(xbuf, lrow, I_PAD, kb, lhi);
        h16v bh = load_b_frag(W0hi, nbase + lrow, I_PAD, kb, lhi);
        h16v bl = load_b_frag(W0lo, nbase + lrow, I_PAD, kb, lhi);
        acc = wmma_f16(a, bh, acc);
        acc = wmma_f16(a, bl, acc);
      }
      s0 = acc;
      f8v mm = m0 * BETA + s0;                  // membrane update
      #pragma unroll
      for (int r = 0; r < 8; ++r) {             // C/D elem: M=r+8*lhi, N=lrow
        float k = mm[r] > 1.0f ? 1.0f : 0.0f;
        k0buf[(r + lhi * 8) * H_DIM + nbase + lrow] = (_Float16)k;
        mm[r] *= (1.0f - k);                    // reset on spike
      }
      m0 = mm;
    }
    __syncthreads();

    // ---- layer 1: s1 = ALPHA*s1 + k0 @ W1^T + k1_prev @ V1^T ----
    {
      const _Float16* k1old = k1buf[t & 1];
      f8v acc = s1 * ALPHA;
      #pragma unroll
      for (int kt = 0; kt < H_DIM / 32; ++kt) {
        int kb = kt * 32;
        h16v a0 = load_a_frag(k0buf, lrow, H_DIM, kb, lhi);
        h16v bh = load_b_frag(W1hi, nbase + lrow, H_DIM, kb, lhi);
        h16v bl = load_b_frag(W1lo, nbase + lrow, H_DIM, kb, lhi);
        acc = wmma_f16(a0, bh, acc);
        acc = wmma_f16(a0, bl, acc);
        h16v a1 = load_a_frag(k1old, lrow, H_DIM, kb, lhi);
        h16v vh = load_b_frag(V1hi, nbase + lrow, H_DIM, kb, lhi);
        h16v vl = load_b_frag(V1lo, nbase + lrow, H_DIM, kb, lhi);
        acc = wmma_f16(a1, vh, acc);
        acc = wmma_f16(a1, vl, acc);
      }
      s1 = acc;
      f8v mm = m1 * BETA + s1;
      _Float16* k1new = k1buf[(t + 1) & 1];
      #pragma unroll
      for (int r = 0; r < 8; ++r) {
        float k = mm[r] > 1.0f ? 1.0f : 0.0f;
        k1new[(r + lhi * 8) * H_DIM + nbase + lrow] = (_Float16)k;
        mm[r] *= (1.0f - k);
      }
      m1 = mm;
    }
    __syncthreads();

    // ---- readout: rp = LAM*rp + (1-LAM)*(k1 @ Wr^T + br); waves 0,1 only
    // (wave-uniform branch: EXEC stays all-ones inside, as WMMA requires) ----
    if (wid < 2) {
      const _Float16* k1new = k1buf[(t + 1) & 1];
      f8v acc = {};
      #pragma unroll
      for (int kt = 0; kt < H_DIM / 32; ++kt) {
        int kb = kt * 32;
        h16v a  = load_a_frag(k1new, lrow, H_DIM, kb, lhi);
        h16v bh = load_b_frag(Wrhi, nbase + lrow, H_DIM, kb, lhi);
        h16v bl = load_b_frag(Wrlo, nbase + lrow, H_DIM, kb, lhi);
        acc = wmma_f16(a, bh, acc);
        acc = wmma_f16(a, bl, acc);
      }
      int o = nbase + lrow;
      #pragma unroll
      for (int r = 0; r < 8; ++r) {
        float v = LAM * rp[r] + (1.0f - LAM) * (acc[r] + brv);
        rp[r] = v;
        if (o < O_DIM)
          out[((size_t)(bbase + r + lhi * 8) * T_STEPS + t) * O_DIM + o] = v;
      }
    }
    // next-iteration xraw writes target the other parity buffer; k1buf reuse
    // is fenced by the barriers inside the next iteration.
  }
}

// ---------------------------------------------------------------------------
extern "C" void kernel_launch(void* const* d_in, const int* in_sizes, int n_in,
                              void* d_out, int out_size, void* d_ws, size_t ws_size,
                              hipStream_t stream) {
  (void)in_sizes; (void)n_in; (void)out_size; (void)ws_size;

  const float* x  = (const float*)d_in[0];   // (512,100,700)
  const float* W0 = (const float*)d_in[1];   // (128,700)
  const float* W1 = (const float*)d_in[2];   // (128,128)
  const float* V1 = (const float*)d_in[3];   // (128,128)
  const float* Wr = (const float*)d_in[4];   // (20,128)
  const float* br = (const float*)d_in[5];   // (20,)
  float* out = (float*)d_out;                // (512,100,20)

  // Workspace layout (f16): ~496 KB total.
  _Float16* ws = (_Float16*)d_ws;
  const size_t n0 = (size_t)H_DIM * I_PAD;   // 90112
  const size_t nh = (size_t)H_DIM * H_DIM;   // 16384
  const size_t nr = (size_t)O_PAD * H_DIM;   //  4096
  _Float16* W0hi = ws;
  _Float16* W0lo = W0hi + n0;
  _Float16* W1hi = W0lo + n0;
  _Float16* W1lo = W1hi + nh;
  _Float16* V1hi = W1lo + nh;
  _Float16* V1lo = V1hi + nh;
  _Float16* Wrhi = V1lo + nh;
  _Float16* Wrlo = Wrhi + nr;

  auto split = [&](const float* s, _Float16* h, _Float16* l,
                   int rs, int cs, int rd, int cd) {
    int n = rd * cd;
    split_f16<<<(n + 255) / 256, 256, 0, stream>>>(s, h, l, rs, cs, rd, cd);
  };
  split(W0, W0hi, W0lo, H_DIM, I_DIM, H_DIM, I_PAD);
  split(W1, W1hi, W1lo, H_DIM, H_DIM, H_DIM, H_DIM);
  split(V1, V1hi, V1lo, H_DIM, H_DIM, H_DIM, H_DIM);
  split(Wr, Wrhi, Wrlo, O_DIM, H_DIM, O_PAD, H_DIM);

  snn_fused<<<B_TOT / TILE_B, 256, 0, stream>>>(
      x, W0hi, W0lo, W1hi, W1lo, V1hi, V1lo, Wrhi, Wrlo, br, out);
}
